// FactorGraphNetwork_3822520893461
// MI455X (gfx1250) — matrix operand order repaired and checked by reading, not compiled, not run on hardware
//
#include <hip/hip_runtime.h>
#include <hip/hip_bf16.h>

// FactorGraphNetwork on MI455X (gfx1250, wave32).
// Layer1/Layer2 on v_wmma_f32_16x16x32_f16; STE masks folded into pre-swizzled
// f16 weight tiles (WMMA B-operand layout). Each wave owns TWO 16-row batch
// tiles so every B-operand load feeds two WMMAs; per-factor scalars are packed
// into one float4 stream; the layer-3 lane reduction is a single post-loop
// butterfly.

typedef __attribute__((ext_vector_type(16))) _Float16 v16h;
typedef __attribute__((ext_vector_type(8)))  float    v8f;
typedef __attribute__((ext_vector_type(4)))  float    v4f;

#define NK 64          // factors
#define NV 64          // input dim
#define NH 16          // hidden dim
#define LOGIT_TH (-0.84729786f)   // ln(0.3/0.7): sigmoid(r) > 0.3  <=>  r > this

// ---------------------------------------------------------------------------
// Prep: hard-threshold masks/weights, fold mask into W1, cast to f16, and
// swizzle W1m / W2 into the ISA 16-bit B-matrix (32x16) per-lane layout:
//   element (K,N):  lane = N + 16*((K>>3)&1),  elem = (K&7) + 8*(K>>4)
// W2 has real K=16 only; K=16..31 (elems 8..15) are zero-padded.
// Also packs per-(k,lane) metadata {b1[n], b2[n], wk*W3[k,n], wk*b3k}.
// ---------------------------------------------------------------------------
__global__ __launch_bounds__(64) void fgn_prep(
    const float* __restrict__ raw_masks,   // [K,V]
    const float* __restrict__ raw_weights, // [K]
    const float* __restrict__ W1,          // [K,V,H]
    const float* __restrict__ b1,          // [K,H]
    const float* __restrict__ W2,          // [K,H,H]
    const float* __restrict__ b2,          // [K,H]
    const float* __restrict__ W3,          // [K,H,1]
    const float* __restrict__ b3,          // [K,1]
    _Float16* __restrict__ wsW1,           // [K][2 tiles][32 lanes][16 elems]
    _Float16* __restrict__ wsW2,           // [K][32 lanes][16 elems]
    float* __restrict__ wsMeta)            // [K][32 lanes][4]
{
    const int k = blockIdx.x;    // 0..63
    const int t = threadIdx.x;   // 0..63

    // ---- W1m swizzle: this thread owns input row v = t ----
    const float m = (raw_masks[k*NV + t] > LOGIT_TH) ? 1.0f : 0.0f;
    const int tileIdx = t >> 5;          // which K=32 tile of V
    const int kk      = t & 31;          // K within tile
    const int laneHi  = 16 * ((kk >> 3) & 1);  // +16 for K in 8..15, 24..31
    const int e       = (kk & 7) + 8 * (kk >> 4);
    #pragma unroll
    for (int h = 0; h < NH; ++h) {
        float val = m * W1[(k*NV + t)*NH + h];
        wsW1[(((k*2 + tileIdx)*32) + laneHi + h)*16 + e] = (_Float16)val;
    }

    // ---- W2 swizzle (K padded to 32 with zeros) ----
    if (t < 16) {
        const int g = t;  // K index (0..15)
        #pragma unroll
        for (int h = 0; h < NH; ++h)
            wsW2[(k*32 + h + 16*(g >> 3))*16 + (g & 7)] = (_Float16)W2[(k*NH + g)*NH + h];
    } else if (t < 32) {
        #pragma unroll
        for (int e2 = 8; e2 < 16; ++e2) {
            wsW2[(k*32 + t - 16)*16 + e2] = (_Float16)0.0f;
            wsW2[(k*32 + t)*16 + e2] = (_Float16)0.0f;
        }
    }

    // ---- packed per-(k,lane) metadata (replicated for lanes 16..31) ----
    if (t < 32) {
        const int n = t & 15;
        const float wk = (raw_weights[k] > LOGIT_TH) ? 1.0f : 0.0f;
        v4f meta = { b1[k*NH + n],
                     b2[k*NH + n],
                     wk * W3[k*NH + n],   // [K,H,1] flat
                     wk * b3[k] };
        *(v4f*)(wsMeta + (k*32 + t)*4) = meta;
    }
}

// Build the f16 A-operand pair (two K=32 tiles of V=64) for one 16-row tile.
// ISA A layout (16-bit A 16x32): lane l holds row M=l&15;
//   elem e(0..7) -> K = e + 8*(l>=16); elem e(8..15) -> K = 16+(e-8)+8*(l>=16)
__device__ __forceinline__ void build_a(const float* __restrict__ xr,
                                        v16h& a0, v16h& a1) {
    v4f c0 = *(const v4f*)(xr + 0);
    v4f c1 = *(const v4f*)(xr + 4);
    v4f c2 = *(const v4f*)(xr + 16);
    v4f c3 = *(const v4f*)(xr + 20);
    v4f d0 = *(const v4f*)(xr + 32);
    v4f d1 = *(const v4f*)(xr + 36);
    v4f d2 = *(const v4f*)(xr + 48);
    v4f d3 = *(const v4f*)(xr + 52);
    #pragma unroll
    for (int i = 0; i < 4; ++i) {
        a0[i]      = (_Float16)c0[i];
        a0[4 + i]  = (_Float16)c1[i];
        a0[8 + i]  = (_Float16)c2[i];
        a0[12 + i] = (_Float16)c3[i];
        a1[i]      = (_Float16)d0[i];
        a1[4 + i]  = (_Float16)d1[i];
        a1[8 + i]  = (_Float16)d2[i];
        a1[12 + i] = (_Float16)d3[i];
    }
}

// ---------------------------------------------------------------------------
// Main: one wave per TWO 16-row batch tiles (M=32); loop over K=64 factors.
//   layer1: 2x wmma per row-tile (V=64 -> two K=32 steps), B-operands shared
//   layer2: 1x wmma per row-tile (K padded 16->32)
//   layer3: per-lane FMA accumulation; single shfl_xor butterfly after loop
// ISA C/D layout: vgpr r, lane l -> (M = r + 8*(l>=16), N = l&15)
// ---------------------------------------------------------------------------
__global__ __launch_bounds__(256) void fgn_main(
    const float*    __restrict__ x,      // [B, 64]
    const _Float16* __restrict__ wsW1,
    const _Float16* __restrict__ wsW2,
    const float*    __restrict__ wsMeta,
    float*          __restrict__ out)    // [B]
{
    __shared__ float lds[8 * 2 * 16 * 18];  // two 16x16 f32 tiles per wave

    const int lane    = threadIdx.x & 31;
    const int wave    = threadIdx.x >> 5;
    const int tile0   = (blockIdx.x * 8 + wave) * 2;  // tiles tile0, tile0+1
    const int rowBase = tile0 * 16;                   // 32 rows per wave
    const int n       = lane & 15;                    // N / column index
    const int hi      = lane >> 4;                    // 0 or 1 (upper half)
    float* ldsA = lds + wave * (2 * 16 * 18);
    float* ldsB = ldsA + 16 * 18;

    // ---- f16 A-operands for both x row tiles ----
    v16h a0A, a1A, a0B, a1B;
    build_a(x + (rowBase +      n) * NV + hi * 8, a0A, a1A);
    build_a(x + (rowBase + 16 + n) * NV + hi * 8, a0B, a1B);

    v8f outA = {0.f,0.f,0.f,0.f,0.f,0.f,0.f,0.f};
    v8f outB = {0.f,0.f,0.f,0.f,0.f,0.f,0.f,0.f};
    float cterm = 0.0f;   // sum_k wk * b3k  (uniform across lanes)

    for (int k = 0; k < NK; ++k) {
        // ---- shared B-operands + packed scalars for this factor ----
        v16h bw0 = *(const v16h*)(wsW1 + ((k*2 + 0)*32 + lane)*16);
        v16h bw1 = *(const v16h*)(wsW1 + ((k*2 + 1)*32 + lane)*16);
        v16h w2t = *(const v16h*)(wsW2 + (k*32 + lane)*16);
        v4f  meta = *(const v4f*)(wsMeta + (k*32 + lane)*4);
        const float bias1 = meta[0];
        const float bias2 = meta[1];
        const float ww3   = meta[2];   // wk * W3[k,n]
        cterm += meta[3];              // wk * b3k

        // ---- layer 1 on both row tiles ----
        v8f accA = {};
        accA = __builtin_amdgcn_wmma_f32_16x16x32_f16(false, a0A, false, bw0,
                                                      (short)0, accA, false, false);
        accA = __builtin_amdgcn_wmma_f32_16x16x32_f16(false, a1A, false, bw1,
                                                      (short)0, accA, false, false);
        v8f accB = {};
        accB = __builtin_amdgcn_wmma_f32_16x16x32_f16(false, a0B, false, bw0,
                                                      (short)0, accB, false, false);
        accB = __builtin_amdgcn_wmma_f32_16x16x32_f16(false, a1B, false, bw1,
                                                      (short)0, accB, false, false);

        // C layout -> plain [m][n] f32 tiles in LDS (stride 18: conflict-free)
        #pragma unroll
        for (int r = 0; r < 8; ++r) {
            ldsA[(r + 8*hi)*18 + n] = fmaxf(accA[r] + bias1, 0.0f);
            ldsB[(r + 8*hi)*18 + n] = fmaxf(accB[r] + bias1, 0.0f);
        }
        asm volatile("s_wait_dscnt 0" ::: "memory");

        // re-read as A-operands (real K=0..15, pad K=16..31 with zeros)
        v16h h1A, h1B;
        {
            const float* sA = ldsA + n*18 + hi*8;
            const float* sB = ldsB + n*18 + hi*8;
            #pragma unroll
            for (int e = 0; e < 8; ++e) {
                h1A[e] = (_Float16)sA[e];
                h1B[e] = (_Float16)sB[e];
            }
            #pragma unroll
            for (int e = 8; e < 16; ++e) {
                h1A[e] = (_Float16)0.0f;
                h1B[e] = (_Float16)0.0f;
            }
        }

        // ---- layer 2 on both row tiles (shared W2 operand) ----
        v8f acc2A = {};
        acc2A = __builtin_amdgcn_wmma_f32_16x16x32_f16(false, h1A, false, w2t,
                                                       (short)0, acc2A, false, false);
        v8f acc2B = {};
        acc2B = __builtin_amdgcn_wmma_f32_16x16x32_f16(false, h1B, false, w2t,
                                                       (short)0, acc2B, false, false);

        // ---- layer 3, deferred: per-lane FMA only (no in-loop shuffles) ----
        #pragma unroll
        for (int r = 0; r < 8; ++r) {
            outA[r] += ww3 * fmaxf(acc2A[r] + bias2, 0.0f);
            outB[r] += ww3 * fmaxf(acc2B[r] + bias2, 0.0f);
        }
    }

    // ---- single butterfly reduction across each 16-lane half ----
    #pragma unroll
    for (int r = 0; r < 8; ++r) {
        float tA = outA[r];
        tA += __shfl_xor(tA, 1, 32);
        tA += __shfl_xor(tA, 2, 32);
        tA += __shfl_xor(tA, 4, 32);
        tA += __shfl_xor(tA, 8, 32);
        outA[r] = tA + cterm;
        float tB = outB[r];
        tB += __shfl_xor(tB, 1, 32);
        tB += __shfl_xor(tB, 2, 32);
        tB += __shfl_xor(tB, 4, 32);
        tB += __shfl_xor(tB, 8, 32);
        outB[r] = tB + cterm;
    }

    // lanes 0 and 16 hold rows 0..7 / 8..15 of each tile; one writer per row
    if (n == 0) {
        float* oA = out + rowBase + 8*hi;
        v4f a0v = {outA[0], outA[1], outA[2], outA[3]};
        v4f a1v = {outA[4], outA[5], outA[6], outA[7]};
        *(v4f*)(oA + 0) = a0v;
        *(v4f*)(oA + 4) = a1v;
        float* oB = out + rowBase + 16 + 8*hi;
        v4f b0v = {outB[0], outB[1], outB[2], outB[3]};
        v4f b1v = {outB[4], outB[5], outB[6], outB[7]};
        *(v4f*)(oB + 0) = b0v;
        *(v4f*)(oB + 4) = b1v;
    }
}

extern "C" void kernel_launch(void* const* d_in, const int* in_sizes, int n_in,
                              void* d_out, int out_size, void* d_ws, size_t ws_size,
                              hipStream_t stream) {
    const float* x           = (const float*)d_in[0];
    const float* raw_masks   = (const float*)d_in[1];
    const float* raw_weights = (const float*)d_in[2];
    const float* W1          = (const float*)d_in[3];
    const float* b1          = (const float*)d_in[4];
    const float* W2          = (const float*)d_in[5];
    const float* b2          = (const float*)d_in[6];
    const float* W3          = (const float*)d_in[7];
    const float* b3          = (const float*)d_in[8];

    char* ws = (char*)d_ws;
    _Float16* wsW1  = (_Float16*)(ws);                // 64*2*32*16 h = 128 KB
    _Float16* wsW2  = (_Float16*)(ws + (128 << 10));  // 64*32*16  h =  64 KB
    float*    wsMeta = (float*)(ws + (192 << 10));    // 64*32*4   f =  32 KB

    fgn_prep<<<NK, 64, 0, stream>>>(raw_masks, raw_weights, W1, b1, W2, b2, W3, b3,
                                    wsW1, wsW2, wsMeta);

    // 65536 rows / 32 per wave = 2048 waves; 8 waves per block -> 256 blocks
    fgn_main<<<256, 256, 0, stream>>>(x, wsW1, wsW2, wsMeta, (float*)d_out);
}
